// BilstmCrfModel_1073741824660
// MI455X (gfx1250) — matrix operand order repaired and checked by reading, not compiled, not run on hardware
//
#include <hip/hip_runtime.h>
#include <hip/hip_bf16.h>

// ---------------------------------------------------------------------------
// BiLSTM-CRF for MI455X (gfx1250, wave32, WMMA, async global->LDS)
// B=32, T=512, V=50000, E=256, H=512 (4H=2048), L=32
// ---------------------------------------------------------------------------

#define BB 32
#define TT 512
#define EE 256
#define HH 512
#define GG 2048   // 4*H
#define LL 32
#define BT (BB * TT)   // 16384

typedef __attribute__((ext_vector_type(16))) _Float16 v16h;
typedef __attribute__((ext_vector_type(8)))  float    v8f;

// Build a v16h A/B fragment from two 16-byte chunks (LDS or global).
__device__ __forceinline__ v16h frag2(const _Float16* p0, const _Float16* p1) {
    union { uint4 q[2]; v16h h; } u;
    u.q[0] = *(const uint4*)p0;
    u.q[1] = *(const uint4*)p1;
    return u.h;
}
// A fragment, 16-bit 16x32 layout: lane half 'hf' -> elems 0..7 = K k0..k0+7,
// elems 8..15 = K k0+16..k0+23 (k0 = kc*32 + hf*8).  K contiguous in row.
__device__ __forceinline__ v16h a_frag(const _Float16* row, int k0) {
    return frag2(row + k0, row + k0 + 16);
}
// B fragment from repacked weights: 16 f16 contiguous per lane (32 bytes).
__device__ __forceinline__ v16h b_frag(const _Float16* p) {
    return frag2(p, p + 8);
}
__device__ __forceinline__ v8f wmma16(v16h a, v16h b, v8f c) {
    return __builtin_amdgcn_wmma_f32_16x16x32_f16(
        /*neg_a=*/false, a, /*neg_b=*/false, b,
        /*c_mod=*/(short)0, c, /*reuse_a=*/false, /*reuse_b=*/false);
}
__device__ __forceinline__ float sigm(float x) { return 1.0f / (1.0f + __expf(-x)); }

// gfx1250 async global->LDS copy (16B per lane), tracked by ASYNCcnt.
__device__ __forceinline__ void async_g2l_b128(unsigned lds_off, const void* gptr) {
    unsigned long long ga = (unsigned long long)gptr;
    asm volatile("global_load_async_to_lds_b128 %0, %1, off"
                 :: "v"(lds_off), "v"(ga) : "memory");
}
__device__ __forceinline__ void wait_asynccnt0() {
    asm volatile("s_wait_asynccnt 0x0" ::: "memory");
}

// ---------------------------------------------------------------------------
// K0: repack f32 weight [K][N] into WMMA-B fragment order (f16):
//   dst[((kc*NT + nt)*32 + lane)*16 + e] = W[(kc*32 + (lane>>4)*8 + (e&7) + (e>>3)*16)*N + nt*16 + (lane&15)]
// ---------------------------------------------------------------------------
__global__ void repack_w(const float* __restrict__ W, _Float16* __restrict__ Wp,
                         int K, int N) {
    int d = blockIdx.x * blockDim.x + threadIdx.x;
    int total = K * N;
    if (d >= total) return;
    int e    = d & 15;
    int lane = (d >> 4) & 31;
    int rest = d >> 9;
    int NT   = N >> 4;
    int nt   = rest % NT;
    int kc   = rest / NT;
    int k = kc * 32 + (lane >> 4) * 8 + (e & 7) + (e >> 3) * 16;
    int n = nt * 16 + (lane & 15);
    Wp[d] = (_Float16)W[k * N + n];
}

// ---------------------------------------------------------------------------
// K1: embedding gather -> x (f16), row-major [bt][E]
// ---------------------------------------------------------------------------
__global__ void embed_gather(const int* __restrict__ text, const float* __restrict__ emb,
                             _Float16* __restrict__ x) {
    int bt = blockIdx.x;
    int e  = threadIdx.x;
    int tok = text[bt];
    x[bt * EE + e] = (_Float16)emb[tok * EE + e];
}

// ---------------------------------------------------------------------------
// K2: Gx = x @ Wx  (16384x256 @ 256x2048) per direction, f16 out.
// Gx is stored TIME-MAJOR: Gx[(t*32 + b)*GG + col] so each timestep's slice is
// one contiguous 128 KB block for the recurrent kernel.
// Block: 256 thr (8 waves). Block tile 32 rows x 512 cols; wave = 2 Mtiles x 4 Ntiles.
// A tile is staged with gfx1250 async global->LDS (ASYNCcnt).
// ---------------------------------------------------------------------------
#define AS_STRIDE (EE + 8)
__global__ __launch_bounds__(256) void gemm_gx(const _Float16* __restrict__ x,
                                               const _Float16* __restrict__ WxpF,
                                               const _Float16* __restrict__ WxpB,
                                               _Float16* __restrict__ GxF,
                                               _Float16* __restrict__ GxB) {
    const _Float16* Wp = blockIdx.z ? WxpB : WxpF;
    _Float16*       Gx = blockIdx.z ? GxB  : GxF;
    int mblk = blockIdx.x;               // 512 blocks of 32 rows
    int nblk = blockIdx.y;               // 4 blocks of 512 cols

    __shared__ _Float16 As[32][AS_STRIDE];
    int tid = threadIdx.x;
    {   // async-stage 32x256 f16 A tile (whole K): 256 lanes x 4 x 16B
        int r  = tid >> 3;
        int c0 = (tid & 7) * 32;
        const _Float16* g = x + (size_t)(mblk * 32 + r) * EE + c0;
#pragma unroll
        for (int q = 0; q < 4; ++q) {
            unsigned lds_off = (unsigned)(uintptr_t)(&As[r][c0 + 8 * q]);
            async_g2l_b128(lds_off, g + 8 * q);
        }
    }
    wait_asynccnt0();
    __syncthreads();

    int w = tid >> 5, lane = tid & 31;
    int hf = lane >> 4, lc = lane & 15;
    int ntbase = nblk * 32 + w * 4;      // 128 N-tiles total

    v8f acc[2][4] = {};
#pragma unroll
    for (int kc = 0; kc < EE / 32; ++kc) {
        int k0 = kc * 32 + hf * 8;
        v16h a0 = a_frag(&As[lc][0],      k0);
        v16h a1 = a_frag(&As[16 + lc][0], k0);
#pragma unroll
        for (int nt = 0; nt < 4; ++nt) {
            v16h bf = b_frag(Wp + ((size_t)(kc * 128 + ntbase + nt) * 32 + lane) * 16);
            acc[0][nt] = wmma16(a0, bf, acc[0][nt]);
            acc[1][nt] = wmma16(a1, bf, acc[1][nt]);
        }
    }
#pragma unroll
    for (int mt = 0; mt < 2; ++mt)
#pragma unroll
        for (int nt = 0; nt < 4; ++nt) {
            int col = (ntbase + nt) * 16 + lc;
#pragma unroll
            for (int i = 0; i < 8; ++i) {
                int row = mblk * 32 + mt * 16 + hf * 8 + i;   // row = b*T + t
                int b = row >> 9;
                int t = row & (TT - 1);
                Gx[((size_t)t * BB + b) * GG + col] = (_Float16)acc[mt][nt][i];
            }
        }
}

// ---------------------------------------------------------------------------
// K3: persistent recurrent LSTM. grid.x = 2 (dir), 1024 threads = 32 waves.
// LDS: z (32 x 2052 f32) + h (32 x 520 f16).  Wh streamed from L2 (repacked).
// Next timestep's Gx slice (contiguous 128 KB) is prefetched each step.
// ---------------------------------------------------------------------------
#define ZS 2052
#define HS 520
#define LSTM_SMEM (32 * ZS * 4 + 32 * HS * 2)
__global__ __launch_bounds__(1024) void lstm_rec(const _Float16* __restrict__ GxF,
                                                 const _Float16* __restrict__ GxB,
                                                 const _Float16* __restrict__ WhpF,
                                                 const _Float16* __restrict__ WhpB,
                                                 const float* __restrict__ bF,
                                                 const float* __restrict__ bB,
                                                 const int* __restrict__ text,
                                                 _Float16* __restrict__ hsF,
                                                 _Float16* __restrict__ hsB) {
    extern __shared__ char smem[];
    float*    z    = (float*)smem;
    _Float16* hbuf = (_Float16*)(smem + 32 * ZS * 4);

    int dir = blockIdx.x;
    const _Float16* Gx   = dir ? GxB  : GxF;
    const _Float16* Whp  = dir ? WhpB : WhpF;
    const float*    bias = dir ? bB   : bF;
    _Float16*       hs   = dir ? hsB  : hsF;

    int tid = threadIdx.x, w = tid >> 5, lane = tid & 31;
    int hf = lane >> 4, lc = lane & 15;
    int ntbase = w * 4;                       // wave covers cols [w*64, w*64+64)

    for (int i = tid; i < 32 * HS; i += 1024) hbuf[i] = (_Float16)0.0f;

    float c[16];
#pragma unroll
    for (int k = 0; k < 16; ++k) c[k] = 0.0f;

    float bcache[4];
#pragma unroll
    for (int nt = 0; nt < 4; ++nt) bcache[nt] = bias[(ntbase + nt) * 16 + lc];

    // gate-phase ownership: thread owns 16 (b,j) pairs, all in one batch row
    int pbase = tid * 16;
    int gb = pbase >> 9;                      // batch index
    int gj = pbase & 511;                     // hidden base (multiple of 16)

    __syncthreads();

    for (int s = 0; s < TT; ++s) {
        int t = dir ? (TT - 1 - s) : s;

        // prefetch next timestep's contiguous Gx slice (128 KB = 1024 x 128B)
        if (s + 1 < TT) {
            int tn = dir ? (t - 1) : (t + 1);
            const char* pf = (const char*)(Gx + (size_t)tn * BB * GG) + tid * 128;
            __builtin_prefetch(pf, 0, 0);
        }

        // ---- GEMM: z = h @ Wh  (32 x 2048, K = 512) ----
        v8f acc[2][4] = {};
#pragma unroll 4
        for (int kc = 0; kc < HH / 32; ++kc) {
            int k0 = kc * 32 + hf * 8;
            v16h a0 = a_frag(&hbuf[lc * HS],        k0);
            v16h a1 = a_frag(&hbuf[(16 + lc) * HS], k0);
#pragma unroll
            for (int nt = 0; nt < 4; ++nt) {
                v16h bf = b_frag(Whp + ((size_t)(kc * 128 + ntbase + nt) * 32 + lane) * 16);
                acc[0][nt] = wmma16(a0, bf, acc[0][nt]);
                acc[1][nt] = wmma16(a1, bf, acc[1][nt]);
            }
        }
        // z = acc + Gx[t] + bias, stage into LDS for the gate phase
        const _Float16* gx_t = Gx + (size_t)t * BB * GG;
#pragma unroll
        for (int mt = 0; mt < 2; ++mt)
#pragma unroll
            for (int nt = 0; nt < 4; ++nt) {
                int col = (ntbase + nt) * 16 + lc;
#pragma unroll
                for (int i = 0; i < 8; ++i) {
                    int b = mt * 16 + hf * 8 + i;
                    float g = (float)gx_t[(size_t)b * GG + col];
                    z[b * ZS + col] = acc[mt][nt][i] + g + bcache[nt];
                }
            }
        __syncthreads();

        // ---- gates + state update ----
        bool m = text[gb * TT + t] != 0;
#pragma unroll 4
        for (int k = 0; k < 16; ++k) {
            int j = gj + k;
            float zi = z[gb * ZS + j];
            float zf = z[gb * ZS + HH + j];
            float zg = z[gb * ZS + 2 * HH + j];
            float zo = z[gb * ZS + 3 * HH + j];
            float ig = sigm(zi), fg = sigm(zf), og = sigm(zo);
            float gg = tanhf(zg);
            float cn = fg * c[k] + ig * gg;
            float hn = og * tanhf(cn);
            float hold = (float)hbuf[gb * HS + j];
            float hv = m ? hn : hold;
            c[k] = m ? cn : c[k];
            hbuf[gb * HS + j] = (_Float16)hv;
            hs[((size_t)gb * TT + t) * HH + j] = (_Float16)hv;
        }
        __syncthreads();
    }
}

// ---------------------------------------------------------------------------
// K4: logits = [hs_f ; hs_b] @ Wd + bd   (16384x1024 @ 1024x32)
// Block: 256 thr (8 waves); wave = 1 Mtile x 2 Ntiles -> block covers 128 rows.
// ---------------------------------------------------------------------------
__global__ __launch_bounds__(256) void decode_gemm(const _Float16* __restrict__ hsF,
                                                   const _Float16* __restrict__ hsB,
                                                   const _Float16* __restrict__ Wdp,
                                                   const float* __restrict__ bd,
                                                   float* __restrict__ logits) {
    int tid = threadIdx.x, w = tid >> 5, lane = tid & 31;
    int hf = lane >> 4, lc = lane & 15;
    int row0 = blockIdx.x * 128 + w * 16;

    v8f acc[2] = {};
#pragma unroll 4
    for (int kc = 0; kc < 32; ++kc) {
        const _Float16* src = (kc < 16) ? hsF : hsB;
        int kk = (kc & 15) * 32 + hf * 8;
        const _Float16* ap = src + (size_t)(row0 + lc) * HH + kk;
        v16h a = frag2(ap, ap + 16);
#pragma unroll
        for (int nt = 0; nt < 2; ++nt) {
            v16h bf = b_frag(Wdp + ((size_t)(kc * 2 + nt) * 32 + lane) * 16);
            acc[nt] = wmma16(a, bf, acc[nt]);
        }
    }
#pragma unroll
    for (int nt = 0; nt < 2; ++nt) {
        int col = nt * 16 + lc;
#pragma unroll
        for (int i = 0; i < 8; ++i) {
            int row = row0 + hf * 8 + i;
            logits[(size_t)row * LL + col] = acc[nt][i] + bd[col];
        }
    }
}

// ---------------------------------------------------------------------------
// K5: Viterbi. One wave (32 threads = 32 states) per sequence.
// ---------------------------------------------------------------------------
__global__ __launch_bounds__(32) void viterbi(const float* __restrict__ logits,
                                              const int* __restrict__ text,
                                              const float* __restrict__ trans,
                                              unsigned char* __restrict__ bp,
                                              int* __restrict__ out) {
    int b = blockIdx.x;
    int j = threadIdx.x;
    __shared__ float tr[LL * LL];
    __shared__ float sc[LL];
    for (int i = j; i < LL * LL; i += LL) tr[i] = trans[i];
    sc[j] = logits[((size_t)b * TT + 0) * LL + j];
    __syncthreads();

    for (int t = 1; t < TT; ++t) {
        float best = sc[0] + tr[0 * LL + j];
        int   bi = 0;
#pragma unroll 8
        for (int i = 1; i < LL; ++i) {
            float v = sc[i] + tr[i * LL + j];
            if (v > best) { best = v; bi = i; }   // first-max like jnp.argmax
        }
        bool m = text[b * TT + t] != 0;
        float nv = best + logits[((size_t)b * TT + t) * LL + j];
        __syncthreads();
        if (m) sc[j] = nv;
        bp[((size_t)b * TT + t) * LL + j] = m ? (unsigned char)bi : (unsigned char)j;
        __syncthreads();
    }
    if (j == 0) {
        float best = sc[0]; int tag = 0;
        for (int i = 1; i < LL; ++i) if (sc[i] > best) { best = sc[i]; tag = i; }
        out[b * TT + (TT - 1)] = (text[b * TT + (TT - 1)] != 0) ? tag : 0;
        for (int t = TT - 2; t >= 0; --t) {
            tag = (int)bp[((size_t)b * TT + (t + 1)) * LL + tag];
            out[b * TT + t] = (text[b * TT + t] != 0) ? tag : 0;
        }
    }
}

// ---------------------------------------------------------------------------
// Host launcher
// ---------------------------------------------------------------------------
static inline size_t align_up(size_t x) { return (x + 255) & ~(size_t)255; }

extern "C" void kernel_launch(void* const* d_in, const int* in_sizes, int n_in,
                              void* d_out, int out_size, void* d_ws, size_t ws_size,
                              hipStream_t stream) {
    const int*   text = (const int*)  d_in[0];
    const float* emb  = (const float*)d_in[1];
    const float* Wx_f = (const float*)d_in[2];
    const float* Wh_f = (const float*)d_in[3];
    const float* b_f  = (const float*)d_in[4];
    const float* Wx_b = (const float*)d_in[5];
    const float* Wh_b = (const float*)d_in[6];
    const float* b_b  = (const float*)d_in[7];
    const float* Wd   = (const float*)d_in[8];
    const float* bd   = (const float*)d_in[9];
    const float* trans= (const float*)d_in[10];
    int* out = (int*)d_out;

    // ---- workspace layout ----
    char* p = (char*)d_ws;
    size_t off = 0;
    auto take = [&](size_t bytes) { char* r = p + off; off = align_up(off + bytes); return r; };

    _Float16* x_f16  = (_Float16*)take((size_t)BT * EE * 2);
    _Float16* Wxp_f  = (_Float16*)take((size_t)EE * GG * 2);
    _Float16* Wxp_b  = (_Float16*)take((size_t)EE * GG * 2);
    _Float16* Whp_f  = (_Float16*)take((size_t)HH * GG * 2);
    _Float16* Whp_b  = (_Float16*)take((size_t)HH * GG * 2);
    _Float16* Wdp    = (_Float16*)take((size_t)(2 * HH) * LL * 2);
    _Float16* Gx_f   = (_Float16*)take((size_t)BT * GG * 2);
    _Float16* Gx_b   = (_Float16*)take((size_t)BT * GG * 2);
    _Float16* hs_f   = (_Float16*)take((size_t)BT * HH * 2);
    _Float16* hs_b   = (_Float16*)take((size_t)BT * HH * 2);
    float*    logits = (float*)   take((size_t)BT * LL * 4);
    unsigned char* bp = (unsigned char*)take((size_t)BT * LL);
    (void)ws_size; (void)n_in; (void)in_sizes; (void)out_size;

    // ---- weight repacking (f32 -> f16 WMMA-B fragment order) ----
    {
        int tot = EE * GG;
        repack_w<<<(tot + 255) / 256, 256, 0, stream>>>(Wx_f, Wxp_f, EE, GG);
        repack_w<<<(tot + 255) / 256, 256, 0, stream>>>(Wx_b, Wxp_b, EE, GG);
        tot = HH * GG;
        repack_w<<<(tot + 255) / 256, 256, 0, stream>>>(Wh_f, Whp_f, HH, GG);
        repack_w<<<(tot + 255) / 256, 256, 0, stream>>>(Wh_b, Whp_b, HH, GG);
        tot = 2 * HH * LL;
        repack_w<<<(tot + 255) / 256, 256, 0, stream>>>(Wd, Wdp, 2 * HH, LL);
    }

    // ---- embedding ----
    embed_gather<<<BT, EE, 0, stream>>>(text, emb, x_f16);

    // ---- Gx = x @ Wx (both directions), time-major output ----
    gemm_gx<<<dim3(BT / 32, GG / 512, 2), 256, 0, stream>>>(x_f16, Wxp_f, Wxp_b, Gx_f, Gx_b);

    // ---- persistent recurrent LSTM (fwd + bwd in parallel) ----
    lstm_rec<<<2, 1024, LSTM_SMEM, stream>>>(Gx_f, Gx_b, Whp_f, Whp_b, b_f, b_b,
                                             text, hs_f, hs_b);

    // ---- decode GEMM ----
    decode_gemm<<<BT / 128, 256, 0, stream>>>(hs_f, hs_b, Wdp, bd, logits);

    // ---- Viterbi ----
    viterbi<<<BB, LL, 0, stream>>>(logits, text, trans, bp, out);
}
